// AttentionHead_53137335386849
// MI455X (gfx1250) — compile-verified, matrix operand order checked by reading, not run on hardware
//
#include <hip/hip_runtime.h>

// Problem constants (match reference)
#define B_   4
#define T_   4096
#define DIN  1024
#define DOUT 64

typedef __bf16 bf16;
typedef bf16  v16bf __attribute__((ext_vector_type(16)));
typedef bf16  v8bf  __attribute__((ext_vector_type(8)));
typedef float v8f   __attribute__((ext_vector_type(8)));
typedef int   v4i   __attribute__((ext_vector_type(4)));

#define AS1 __attribute__((address_space(1)))
#define AS3 __attribute__((address_space(3)))

// Async global->LDS copy path (CDNA5 GLOBAL_LOAD_ASYNC_TO_LDS_B128), guarded
// so the file still compiles on toolchains without the builtins.
#if defined(__has_builtin)
#if __has_builtin(__builtin_amdgcn_global_load_async_to_lds_b128) && \
    __has_builtin(__builtin_amdgcn_s_wait_asynccnt)
#define USE_ASYNC_LDS 1
#endif
#endif

__device__ __forceinline__ v8f wmma_bf16(v16bf a, v16bf b, v8f c) {
    // D = A(16x32 bf16) * B(32x16 bf16) + C(16x16 f32)
    return __builtin_amdgcn_wmma_f32_16x16x32_bf16(
        false, a, false, b, (short)0, c, false, false);
}

__device__ __forceinline__ v16bf combine8(v8bf lo, v8bf hi) {
    return __builtin_shufflevector(lo, hi, 0, 1, 2, 3, 4, 5, 6, 7,
                                   8, 9, 10, 11, 12, 13, 14, 15);
}

#if USE_ASYNC_LDS
__device__ __forceinline__ void async_copy16(const bf16* g, bf16* l) {
    __builtin_amdgcn_global_load_async_to_lds_b128(
        (AS1 v4i*)(uintptr_t)g, (AS3 v4i*)(uintptr_t)l, 0, 0);
}
#endif

// ---------------------------------------------------------------------------
// Kernel 0: transpose + convert weights: Wt[m][n][k] = W_m[k][n] as bf16.
// Runs once per launch; 3 * 64 * 1024 elements, bandwidth-trivial.
// ---------------------------------------------------------------------------
__global__ __launch_bounds__(256)
void wt_bf16_kernel(const float* __restrict__ Wq,
                    const float* __restrict__ Wk,
                    const float* __restrict__ Wv,
                    bf16* __restrict__ Wt) {
    const int n = blockIdx.x;                 // output column 0..63
    const int m = blockIdx.y;                 // matrix 0..2
    const float* W = (m == 0) ? Wq : (m == 1) ? Wk : Wv;
    bf16* dst = Wt + ((size_t)m * DOUT + n) * DIN;
    for (int k = threadIdx.x; k < DIN; k += 256)
        dst[k] = (bf16)W[(size_t)k * DOUT + n];
}

// ---------------------------------------------------------------------------
// Kernel 1: fused QKV projection.  X:[B*T, DIN] fp32, Wt:[3,DOUT,DIN] bf16.
// Block = 128 threads (4 waves). Block owns a 16-row M tile; wave w owns
// output columns [16w, 16w+16). bf16 WMMA, f32 accumulate, bf16 store.
// Q is pre-scaled by 1/sqrt(d) = 0.125 at store.
// ---------------------------------------------------------------------------
__global__ __launch_bounds__(128)
void qkv_proj_kernel(const float* __restrict__ x,
                     const bf16*  __restrict__ Wt,
                     bf16* __restrict__ Q,
                     bf16* __restrict__ K,
                     bf16* __restrict__ V) {
    const int lane = threadIdx.x & 31;
    const int wid  = threadIdx.x >> 5;   // 0..3 -> N tile
    const int half = lane >> 4;          // 0/1
    const int lm   = lane & 15;
    const int row0 = blockIdx.x * 16;    // row into flattened [B*T]
    const int col  = wid * 16 + lm;      // output column (B/C lane = column)

    v8f cq = {}; v8f ck = {}; v8f cv = {};
    const float* xr = x + (size_t)(row0 + lm) * DIN;           // A: row per lane
    const bf16*  wq = Wt + ((size_t)0 * DOUT + col) * DIN;     // B: column rows
    const bf16*  wk = Wt + ((size_t)1 * DOUT + col) * DIN;
    const bf16*  wv = Wt + ((size_t)2 * DOUT + col) * DIN;

    for (int kk = 0; kk < DIN; kk += 32) {
        const int o1 = kk + half * 8;        // contraction run 1
        const int o2 = kk + 16 + half * 8;   // contraction run 2
        // A fragment: convert 16 contiguous-run fp32 -> bf16
        v16bf a;
#pragma unroll
        for (int e = 0; e < 8; ++e) {
            a[e]     = (bf16)xr[o1 + e];
            a[8 + e] = (bf16)xr[o2 + e];
        }
        // B fragments: two 16B contiguous loads each from transposed weights
        const v16bf bq = combine8(*(const v8bf*)(wq + o1), *(const v8bf*)(wq + o2));
        const v16bf bk = combine8(*(const v8bf*)(wk + o1), *(const v8bf*)(wk + o2));
        const v16bf bv = combine8(*(const v8bf*)(wv + o1), *(const v8bf*)(wv + o2));
        cq = wmma_bf16(a, bq, cq);
        ck = wmma_bf16(a, bk, ck);
        cv = wmma_bf16(a, bv, cv);
    }

    // C layout: vgpr r, lanes 0-15 -> row r, lanes 16-31 -> row r+8; col = lane&15
#pragma unroll
    for (int r = 0; r < 8; ++r) {
        const size_t idx = (size_t)(row0 + r + 8 * half) * DOUT + col;
        Q[idx] = (bf16)(cq[r] * 0.125f);   // fold softmax scale into Q
        K[idx] = (bf16)ck[r];
        V[idx] = (bf16)cv[r];
    }
}

// ---------------------------------------------------------------------------
// Kernel 2: causal flash attention over bf16 Q,K,V [B, T, 64] (Q pre-scaled).
// Block = (batch, 64-query tile), 4 waves, wave w owns rows [q0b+16w, +16).
// k-loop over 32-key tiles, double-buffered in LDS via async DMA: while tile i
// is consumed by WMMA, tile i+1 is in flight (ASYNCcnt retires in issue order,
// so s_wait_asynccnt(4) guarantees tile i's 4 copies have landed).
// ---------------------------------------------------------------------------
#define KPAD  72                 // padded row (bf16 elems); 144 B = 9*16
#define TILE_ (32 * KPAD)        // one K or V tile in LDS

__global__ __launch_bounds__(128)
void flash_attn_kernel(const bf16* __restrict__ Qg,
                       const bf16* __restrict__ Kg,
                       const bf16* __restrict__ Vg,
                       float* __restrict__ out) {
    __shared__ __align__(16) bf16 Ks[2 * TILE_];
    __shared__ __align__(16) bf16 Vs[2 * TILE_];
    __shared__ __align__(16) bf16 Ps[4 * 16 * 32];   // per-wave P scratch

    const int tid  = threadIdx.x;
    const int lane = tid & 31;
    const int wid  = tid >> 5;
    const int half = lane >> 4;
    const int lm   = lane & 15;
    const int b    = blockIdx.y;
    const int q0b  = blockIdx.x * 64;
    const int q0   = q0b + wid * 16;
    const int nt   = (q0b + 64) / 32;    // tiles, uniform across the block

    const bf16* Qb = Qg + (size_t)b * T_ * DOUT;
    const bf16* Kb = Kg + (size_t)b * T_ * DOUT;
    const bf16* Vb = Vg + (size_t)b * T_ * DOUT;

    const int ckey = tid >> 3;           // 0..15: key row this thread copies
    const int cch  = tid & 7;            // 16B chunk within the 128B row

    // Stage one 32-key K/V tile into LDS buffer `buf` (4 x 16B per thread)
    auto issue_tile = [&](int k0, int buf) {
        const bf16* gK0 = Kb + (size_t)(k0 + ckey) * DOUT + cch * 8;
        const bf16* gV0 = Vb + (size_t)(k0 + ckey) * DOUT + cch * 8;
        bf16* lK0 = &Ks[buf * TILE_ + ckey * KPAD + cch * 8];
        bf16* lV0 = &Vs[buf * TILE_ + ckey * KPAD + cch * 8];
#if USE_ASYNC_LDS
        async_copy16(gK0, lK0);
        async_copy16(gK0 + 16 * DOUT, lK0 + 16 * KPAD);
        async_copy16(gV0, lV0);
        async_copy16(gV0 + 16 * DOUT, lV0 + 16 * KPAD);
#else
        *(v8bf*)lK0 = *(const v8bf*)gK0;
        *(v8bf*)(lK0 + 16 * KPAD) = *(const v8bf*)(gK0 + 16 * DOUT);
        *(v8bf*)lV0 = *(const v8bf*)gV0;
        *(v8bf*)(lV0 + 16 * KPAD) = *(const v8bf*)(gV0 + 16 * DOUT);
#endif
    };

    // Preload Q A-fragments for e-chunks 0-31 and 32-63 (already scaled)
    v16bf aq[2];
#pragma unroll
    for (int c = 0; c < 2; ++c) {
        const bf16* qr = Qb + (size_t)(q0 + lm) * DOUT + 32 * c;
        aq[c] = combine8(*(const v8bf*)(qr + half * 8),
                         *(const v8bf*)(qr + 16 + half * 8));
    }

    v8f o[4] = {v8f{}, v8f{}, v8f{}, v8f{}};
    float mr[8], lr[8];
#pragma unroll
    for (int r = 0; r < 8; ++r) { mr[r] = -1e30f; lr[r] = 0.f; }

    bf16* Pw = &Ps[wid * 16 * 32];

    issue_tile(0, 0);                    // prime the pipeline

    for (int it = 0; it < nt; ++it) {
        const int k0 = it * 32;
        __syncthreads();   // everyone finished reading the buffer we refill next
        if (it + 1 < nt) {
            issue_tile(k0 + 32, (it + 1) & 1);
#if USE_ASYNC_LDS
            __builtin_amdgcn_s_wait_asynccnt(4);   // oldest 4 (tile it) landed
#endif
        } else {
#if USE_ASYNC_LDS
            __builtin_amdgcn_s_wait_asynccnt(0);
#endif
        }
        __syncthreads();   // current buffer fully populated for all waves

        const bf16* Kt = &Ks[(it & 1) * TILE_];
        const bf16* Vt = &Vs[(it & 1) * TILE_];

        // S = (Q*scale) . K^T : two 16-key N tiles, contraction e=64 in 2 chunks
        v8f s0 = {}, s1 = {};
#pragma unroll
        for (int c = 0; c < 2; ++c) {
            const int ek = 32 * c;
            {
                const bf16* rp = Kt + (size_t)(0 + lm) * KPAD + ek;
                const v16bf bb = combine8(*(const v8bf*)(rp + half * 8),
                                          *(const v8bf*)(rp + 16 + half * 8));
                s0 = wmma_bf16(aq[c], bb, s0);
            }
            {
                const bf16* rp = Kt + (size_t)(16 + lm) * KPAD + ek;
                const v16bf bb = combine8(*(const v8bf*)(rp + half * 8),
                                          *(const v8bf*)(rp + 16 + half * 8));
                s1 = wmma_bf16(aq[c], bb, s1);
            }
        }

        // Causal mask (only when the tile can cross the diagonal)
        if (k0 + 31 > q0) {
#pragma unroll
            for (int r = 0; r < 8; ++r) {
                const int q = q0 + r + 8 * half;
                if (k0 + lm > q)      s0[r] = -1e30f;
                if (k0 + 16 + lm > q) s1[r] = -1e30f;
            }
        }

        // Online softmax update; row reductions across the 16-lane half-wave
#pragma unroll
        for (int r = 0; r < 8; ++r) {
            float mt = fmaxf(s0[r], s1[r]);
#pragma unroll
            for (int off = 8; off >= 1; off >>= 1)
                mt = fmaxf(mt, __shfl_xor(mt, off, 32));
            const float mn = fmaxf(mr[r], mt);
            const float al = __expf(mr[r] - mn);
            const float p0 = __expf(s0[r] - mn);
            const float p1 = __expf(s1[r] - mn);
            float rs = p0 + p1;
#pragma unroll
            for (int off = 8; off >= 1; off >>= 1)
                rs += __shfl_xor(rs, off, 32);
            lr[r] = lr[r] * al + rs;
            mr[r] = mn;
            o[0][r] *= al; o[1][r] *= al; o[2][r] *= al; o[3][r] *= al;
            const int qrow = r + 8 * half;
            Pw[qrow * 32 + lm]      = (bf16)p0;   // C-layout -> LDS row-major
            Pw[qrow * 32 + 16 + lm] = (bf16)p1;
        }

        // Re-read P in A-fragment layout (DS ops are in-order within the wave)
        v16bf ap;
        {
            const bf16* pr = &Pw[(size_t)lm * 32];
            ap = combine8(*(const v8bf*)(pr + half * 8),
                          *(const v8bf*)(pr + 16 + half * 8));
        }

        // O += P . V over 4 e-tiles of 16 columns
#pragma unroll
        for (int t = 0; t < 4; ++t) {
            v16bf bv;
#pragma unroll
            for (int e = 0; e < 8; ++e) {
                bv[e]     = Vt[(size_t)(half * 8 + e) * KPAD + 16 * t + lm];
                bv[8 + e] = Vt[(size_t)(16 + half * 8 + e) * KPAD + 16 * t + lm];
            }
            o[t] = wmma_bf16(ap, bv, o[t]);
        }
    }

    // Epilogue: normalize and store fp32 output [B, T, 64]
#pragma unroll
    for (int r = 0; r < 8; ++r) {
        const float inv = 1.0f / lr[r];
        const int q = q0 + r + 8 * half;
        const size_t base = ((size_t)b * T_ + q) * DOUT;
#pragma unroll
        for (int t = 0; t < 4; ++t)
            out[base + 16 * t + lm] = o[t][r] * inv;
    }
}

// ---------------------------------------------------------------------------
extern "C" void kernel_launch(void* const* d_in, const int* in_sizes, int n_in,
                              void* d_out, int out_size, void* d_ws, size_t ws_size,
                              hipStream_t stream) {
    (void)in_sizes; (void)n_in; (void)out_size; (void)ws_size;

    const float* x  = (const float*)d_in[0];
    const float* Wq = (const float*)d_in[1];
    const float* Wk = (const float*)d_in[2];
    const float* Wv = (const float*)d_in[3];

    // Workspace layout (bf16): Wt[3*64*1024] | Q | K | V  (~6.7 MB total)
    bf16* Wt = (bf16*)d_ws;
    bf16* Q  = Wt + (size_t)3 * DOUT * DIN;
    bf16* K  = Q + (size_t)B_ * T_ * DOUT;
    bf16* V  = K + (size_t)B_ * T_ * DOUT;
    float* out = (float*)d_out;

    wt_bf16_kernel<<<dim3(DOUT, 3), 256, 0, stream>>>(Wq, Wk, Wv, Wt);
    qkv_proj_kernel<<<(B_ * T_) / 16, 128, 0, stream>>>(x, Wt, Q, K, V);
    flash_attn_kernel<<<dim3(T_ / 64, B_), 128, 0, stream>>>(Q, K, V, out);
}
